// LinearAttentionLayer_14937896255975
// MI455X (gfx1250) — matrix operand order repaired and checked by reading, not compiled, not run on hardware
//
#include <hip/hip_runtime.h>
#include <stdint.h>

// ---------------------------------------------------------------------------
// LinearAttentionLayer for MI455X (gfx1250, wave32, WMMA bf16 + async-to-LDS)
// B=4, L=8192, D=1024, H=16, Dh=64
// ---------------------------------------------------------------------------

typedef __attribute__((ext_vector_type(16))) __bf16 v16bf;
typedef __attribute__((ext_vector_type(8)))  float  v8f;

union AFrag { uint4 q[2]; v16bf v; };

__device__ __forceinline__ unsigned short f2bf(float f) {
  union { float f; unsigned u; } c; c.f = f;
  unsigned r = c.u + 0x7fffu + ((c.u >> 16) & 1u);   // round-to-nearest-even
  return (unsigned short)(r >> 16);
}
__device__ __forceinline__ float lo2f(unsigned u) { return __uint_as_float(u << 16); }
__device__ __forceinline__ float hi2f(unsigned u) { return __uint_as_float(u & 0xffff0000u); }

// Low 32 bits of a generic pointer to LDS == LDS byte address (ISA 10.2).
__device__ __forceinline__ unsigned lds_a(const void* p) {
  return (unsigned)(unsigned long long)p;
}
// Async DMA: 16B per lane, global -> LDS, tracked by ASYNCcnt.
__device__ __forceinline__ void async_b128(const void* gptr, unsigned lds_off) {
  asm volatile("global_load_async_to_lds_b128 %0, %1, off"
               :: "v"(lds_off), "v"(gptr) : "memory");
}
__device__ __forceinline__ void wait_async_le4() {
  asm volatile("s_wait_asynccnt 0x4" ::: "memory");
}
__device__ __forceinline__ void wait_async_0() {
  asm volatile("s_wait_asynccnt 0x0" ::: "memory");
}

constexpr int Bn = 4, Ln = 8192, Dn = 1024, Hn = 16, DH = 64;
constexpr int Mtot = Bn * Ln;                 // 32768
constexpr int MODE_Q = 0, MODE_K = 1, MODE_V = 2, MODE_O = 3;
constexpr int LDA = 40;                       // LDS row stride (halves); 80B keeps 16B align

// ---------------------------------------------------------------------------
// Small helper kernels
// ---------------------------------------------------------------------------
// fp32 -> bf16, 8 elements / thread
__global__ void cvt_kernel(const float* __restrict__ in, unsigned short* __restrict__ o) {
  int i = (blockIdx.x * 256 + threadIdx.x) * 8;
  float4 f0 = *(const float4*)(in + i);
  float4 f1 = *(const float4*)(in + i + 4);
  uint4 u;
  u.x = (unsigned)f2bf(f0.x) | ((unsigned)f2bf(f0.y) << 16);
  u.y = (unsigned)f2bf(f0.z) | ((unsigned)f2bf(f0.w) << 16);
  u.z = (unsigned)f2bf(f1.x) | ((unsigned)f2bf(f1.y) << 16);
  u.w = (unsigned)f2bf(f1.z) | ((unsigned)f2bf(f1.w) << 16);
  *(uint4*)(o + i) = u;
}

__global__ void zero_kernel(float* __restrict__ p, int n) {
  int i = blockIdx.x * 256 + threadIdx.x;
  if (i < n) p[i] = 0.0f;
}

// KV (f32, [bh][d][e]) -> KVT (bf16, [bh][e][d])
__global__ void kvt_kernel(const float* __restrict__ KV, unsigned short* __restrict__ KVT) {
  int idx = blockIdx.x * 256 + threadIdx.x;              // interpreted as (bh,e,d)
  int bh = idx >> 12, rem = idx & 4095, e = rem >> 6, d = rem & 63;
  KVT[idx] = f2bf(KV[(bh << 12) + (d << 6) + e]);
}

// Ksum[bh][d] = sum_l KpT[bh][d][l]
__global__ __launch_bounds__(256)
void ksum_kernel(const unsigned short* __restrict__ KpT, float* __restrict__ Ksum) {
  __shared__ float red[256];
  int bh = blockIdx.x, tid = threadIdx.x;
  int d = tid >> 2, q = tid & 3;
  const unsigned short* p = KpT + (size_t)bh * (DH * Ln) + (size_t)d * Ln + q * 2048;
  float s = 0.0f;
  for (int i = 0; i < 2048; i += 8) {
    uint4 u = *(const uint4*)(p + i);
    s += lo2f(u.x) + hi2f(u.x) + lo2f(u.y) + hi2f(u.y)
       + lo2f(u.z) + hi2f(u.z) + lo2f(u.w) + hi2f(u.w);
  }
  red[tid] = s;
  __syncthreads();
  if (q == 0) Ksum[bh * DH + d] = red[tid] + red[tid + 1] + red[tid + 2] + red[tid + 3];
}

// ---------------------------------------------------------------------------
// Big GEMM: C(M=32768, N=1024) = act( A(M,1024) @ W^T + bias ), A/W bf16.
// Block tile 128x128, 8 waves as 4x2, wave tile 32x64 (2x4 WMMA 16x16 tiles).
// Double-buffered LDS tiles filled by GLOBAL_LOAD_ASYNC_TO_LDS_B128.
// MODE_Q : bf16 out, feature map, layout (B,H,L,Dh)
// MODE_K : bf16 out, feature map, layout (B,H,Dh,L)   (transposed)
// MODE_V : bf16 out, mask only,   layout (B,H,Dh,L)   (transposed)
// MODE_O : f32 out, bias only,    layout (M,N) row-major (= d_out)
// ---------------------------------------------------------------------------
template <int MODE>
__global__ __launch_bounds__(256)
void gemm_kernel(const unsigned short* __restrict__ Abf,
                 const unsigned short* __restrict__ Wbf,
                 const float* __restrict__ bias,
                 const unsigned char* __restrict__ mask,
                 void* __restrict__ outp) {
  __shared__ unsigned short lA[2][128 * LDA];
  __shared__ unsigned short lB[2][128 * LDA];

  const int m0 = blockIdx.x * 128;
  const int n0 = blockIdx.y * 128;
  const int tid  = threadIdx.x;
  const int lane = tid & 31;
  const int wid  = tid >> 5;
  const int wm = wid >> 1, wn = wid & 1;

  // Per-thread staging coordinates (2 x b128 per tile per array).
  const int lin0 = tid * 8, lin1 = (256 + tid) * 8;
  const int r0 = lin0 >> 5, c0 = lin0 & 31;
  const int r1 = lin1 >> 5, c1 = lin1 & 31;
  const unsigned short* Ar0 = Abf + (size_t)(m0 + r0) * Dn + c0;
  const unsigned short* Ar1 = Abf + (size_t)(m0 + r1) * Dn + c1;
  const unsigned short* Br0 = Wbf + (size_t)(n0 + r0) * Dn + c0;
  const unsigned short* Br1 = Wbf + (size_t)(n0 + r1) * Dn + c1;
  const unsigned la0 = lds_a(&lA[0][r0 * LDA + c0]), la1 = lds_a(&lA[0][r1 * LDA + c1]);
  const unsigned lb0 = lds_a(&lB[0][r0 * LDA + c0]), lb1 = lds_a(&lB[0][r1 * LDA + c1]);
  const unsigned bufstep = 128 * LDA * 2;   // bytes between double buffers

  auto issue = [&](int k0, int buf) {
    unsigned bo = buf ? bufstep : 0;
    async_b128(Ar0 + k0, la0 + bo);
    async_b128(Ar1 + k0, la1 + bo);
    async_b128(Br0 + k0, lb0 + bo);
    async_b128(Br1 + k0, lb1 + bo);
  };

  v8f acc[2][4];
  const v8f vzero = {0.f, 0.f, 0.f, 0.f, 0.f, 0.f, 0.f, 0.f};
#pragma unroll
  for (int i = 0; i < 2; ++i)
#pragma unroll
    for (int j = 0; j < 4; ++j) acc[i][j] = vzero;

  constexpr int NK = Dn / 32;   // 32 k-steps
  issue(0, 0);
  for (int k = 0; k < NK; ++k) {
    const int cur = k & 1;
    if (k + 1 < NK) { issue((k + 1) * 32, cur ^ 1); wait_async_le4(); }
    else            { wait_async_0(); }
    __syncthreads();

    AFrag a[2];
#pragma unroll
    for (int i = 0; i < 2; ++i) {
      int row = wm * 32 + i * 16 + (lane & 15);
      int ch  = (lane >> 4) * 8;
      a[i].q[0] = *(const uint4*)(&lA[cur][row * LDA + ch]);
      a[i].q[1] = *(const uint4*)(&lA[cur][row * LDA + ch + 16]);
    }
    AFrag bf[4];
#pragma unroll
    for (int j = 0; j < 4; ++j) {
      int row = wn * 64 + j * 16 + (lane & 15);
      int ch  = (lane >> 4) * 8;
      bf[j].q[0] = *(const uint4*)(&lB[cur][row * LDA + ch]);
      bf[j].q[1] = *(const uint4*)(&lB[cur][row * LDA + ch + 16]);
    }
#pragma unroll
    for (int i = 0; i < 2; ++i)
#pragma unroll
      for (int j = 0; j < 4; ++j)
        acc[i][j] = __builtin_amdgcn_wmma_f32_16x16x32_bf16(
            false, a[i].v, false, bf[j].v, (short)0, acc[i][j], false, false);
    __syncthreads();   // all waves done reading buf[cur] before it is refilled
  }

  // ---- epilogue ----
  const int cbase = n0 + wn * 64;
#pragma unroll
  for (int i = 0; i < 2; ++i) {
    int mbase = m0 + wm * 32 + i * 16 + 8 * (lane >> 4);
#pragma unroll
    for (int j = 0; j < 4; ++j) {
      int col = cbase + j * 16 + (lane & 15);
      float bv = bias[col];
#pragma unroll
      for (int r = 0; r < 8; ++r) {
        int M = mbase + r;
        float v = acc[i][j][r] + bv;
        if (MODE == MODE_O) {
          ((float*)outp)[(size_t)M * Dn + col] = v;
        } else {
          int b = M >> 13, l = M & (Ln - 1);
          int h = col >> 6, d = col & 63;
          float mv = mask[(size_t)b * Ln + l] ? 1.0f : 0.0f;
          v *= mv;
          if (MODE != MODE_V) v = (v > 0.0f) ? (v + 1.0f) : __expf(v);  // elu+1
          size_t off;
          if (MODE == MODE_Q)
            off = (((size_t)(b * Hn + h)) * Ln + l) * DH + d;           // (B,H,L,Dh)
          else
            off = (((size_t)(b * Hn + h)) * DH + d) * Ln + l;           // (B,H,Dh,L)
          ((unsigned short*)outp)[off] = f2bf(v);
        }
      }
    }
  }
}

// ---------------------------------------------------------------------------
// KV[bh][d][e] += sum_l Kp[l,d] * V[l,e]   via WMMA, L split 8-way.
// A = KpT (64 x L, l-contiguous rows), B = V as K x N from VT rows.
// Per-wave 64x64 accumulation, LDS ds_add_f32 cross-wave reduce, global atomics.
// ---------------------------------------------------------------------------
__global__ __launch_bounds__(256)
void kv_kernel(const unsigned short* __restrict__ KpT,
               const unsigned short* __restrict__ VT,
               float* __restrict__ KV) {
  __shared__ float sKV[DH * DH];
  const int bh = blockIdx.x;
  const int split = blockIdx.y;                 // 0..7
  const int tid = threadIdx.x, lane = tid & 31, wid = tid >> 5;

  for (int i = tid; i < DH * DH; i += 256) sKV[i] = 0.0f;
  __syncthreads();

  const size_t base = (size_t)bh * DH * Ln;
  const unsigned short* Kb = KpT + base;
  const unsigned short* Vb = VT + base;

  v8f acc[4][4];
  const v8f vzero = {0.f, 0.f, 0.f, 0.f, 0.f, 0.f, 0.f, 0.f};
#pragma unroll
  for (int i = 0; i < 4; ++i)
#pragma unroll
    for (int j = 0; j < 4; ++j) acc[i][j] = vzero;

  const int l0base = split * 1024 + wid * 128;  // each wave: 128 l's = 4 k-steps
  for (int ks = 0; ks < 4; ++ks) {
    int l0 = l0base + ks * 32;
    AFrag a[4], b[4];
#pragma unroll
    for (int i = 0; i < 4; ++i) {
      int row = i * 16 + (lane & 15);
      int ca  = l0 + (lane >> 4) * 8;
      a[i].q[0] = *(const uint4*)(Kb + (size_t)row * Ln + ca);
      a[i].q[1] = *(const uint4*)(Kb + (size_t)row * Ln + ca + 16);
      int cb = l0 + (lane >> 4) * 8;
      b[i].q[0] = *(const uint4*)(Vb + (size_t)row * Ln + cb);
      b[i].q[1] = *(const uint4*)(Vb + (size_t)row * Ln + cb + 16);
    }
#pragma unroll
    for (int i = 0; i < 4; ++i)
#pragma unroll
      for (int j = 0; j < 4; ++j)
        acc[i][j] = __builtin_amdgcn_wmma_f32_16x16x32_bf16(
            false, a[i].v, false, b[j].v, (short)0, acc[i][j], false, false);
  }

#pragma unroll
  for (int i = 0; i < 4; ++i)
#pragma unroll
    for (int j = 0; j < 4; ++j)
#pragma unroll
      for (int r = 0; r < 8; ++r) {
        int m = i * 16 + r + 8 * (lane >> 4);
        int n = j * 16 + (lane & 15);
        atomicAdd(&sKV[m * DH + n], acc[i][j][r]);
      }
  __syncthreads();
  for (int i = tid; i < DH * DH; i += 256)
    atomicAdd(&KV[(size_t)bh * (DH * DH) + i], sKV[i]);
}

// ---------------------------------------------------------------------------
// attn kernel: num = Qp @ KV ; den = Qp . Ksum + 1e-6 ; attn = num/den (bf16,(B,L,D))
// Block: one bh, 128 rows of l. 8 waves, wave tile 16 x 64.
// ---------------------------------------------------------------------------
__global__ __launch_bounds__(256)
void attn_kernel(const unsigned short* __restrict__ Qp,
                 const unsigned short* __restrict__ KVT,
                 const float* __restrict__ Ksum,
                 unsigned short* __restrict__ attn) {
  __shared__ unsigned short sKVT[DH * DH];   // 8 KB
  __shared__ float sKsum[DH];
  __shared__ float sDen[128];

  const int bh = blockIdx.y;
  const int l0 = blockIdx.x * 128;
  const int tid = threadIdx.x, lane = tid & 31, wid = tid >> 5;

  for (int i = tid * 8; i < DH * DH; i += 2048)
    *(uint4*)(&sKVT[i]) = *(const uint4*)(KVT + (size_t)bh * (DH * DH) + i);
  if (tid < DH) sKsum[tid] = Ksum[bh * DH + tid];
  __syncthreads();

  const unsigned short* Qb = Qp + (size_t)bh * ((size_t)Ln * DH);

  // denominator for rows 0..127
  if (tid < 128) {
    const unsigned short* qr = Qb + (size_t)(l0 + tid) * DH;
    float s = 0.0f;
#pragma unroll 8
    for (int d = 0; d < DH; d += 2) {
      unsigned u = *(const unsigned*)(qr + d);
      s += lo2f(u) * sKsum[d] + hi2f(u) * sKsum[d + 1];
    }
    sDen[tid] = s + 1e-6f;
  }

  v8f acc[4];
  const v8f vzero = {0.f, 0.f, 0.f, 0.f, 0.f, 0.f, 0.f, 0.f};
#pragma unroll
  for (int j = 0; j < 4; ++j) acc[j] = vzero;

#pragma unroll
  for (int k0 = 0; k0 < DH; k0 += 32) {
    AFrag a;
    int row = l0 + wid * 16 + (lane & 15);
    int ca  = k0 + (lane >> 4) * 8;
    a.q[0] = *(const uint4*)(Qb + (size_t)row * DH + ca);
    a.q[1] = *(const uint4*)(Qb + (size_t)row * DH + ca + 16);
#pragma unroll
    for (int j = 0; j < 4; ++j) {
      AFrag b;
      int er = j * 16 + (lane & 15);
      int cb = k0 + (lane >> 4) * 8;
      b.q[0] = *(const uint4*)(&sKVT[er * DH + cb]);
      b.q[1] = *(const uint4*)(&sKVT[er * DH + cb + 16]);
      acc[j] = __builtin_amdgcn_wmma_f32_16x16x32_bf16(
          false, a.v, false, b.v, (short)0, acc[j], false, false);
    }
  }
  __syncthreads();   // sDen ready

  const int b = bh >> 4, h = bh & 15;
#pragma unroll
  for (int j = 0; j < 4; ++j) {
    int e = j * 16 + (lane & 15);
#pragma unroll
    for (int r = 0; r < 8; ++r) {
      int rowt = wid * 16 + r + 8 * (lane >> 4);
      float v = acc[j][r] / sDen[rowt];
      int l = l0 + rowt;
      attn[((size_t)(b * Ln + l)) * Dn + h * DH + e] = f2bf(v);
    }
  }
}

// ---------------------------------------------------------------------------
// Launcher
// ---------------------------------------------------------------------------
extern "C" void kernel_launch(void* const* d_in, const int* in_sizes, int n_in,
                              void* d_out, int out_size, void* d_ws, size_t ws_size,
                              hipStream_t stream) {
  const float*         x    = (const float*)d_in[0];
  const unsigned char* mask = (const unsigned char*)d_in[1];
  const float* Wq = (const float*)d_in[2]; const float* bq = (const float*)d_in[3];
  const float* Wk = (const float*)d_in[4]; const float* bk = (const float*)d_in[5];
  const float* Wv = (const float*)d_in[6]; const float* bv = (const float*)d_in[7];
  const float* Wo = (const float*)d_in[8]; const float* bo = (const float*)d_in[9];

  char* ws = (char*)d_ws;
  const size_t WSZ   = (size_t)Dn * Dn * 2;            // 2 MB bf16 weights
  const size_t ACTSZ = (size_t)Bn * Hn * Ln * DH * 2;  // 64 MB bf16 activations
  size_t off = 0;
  unsigned short* wq_bf = (unsigned short*)(ws + off); off += WSZ;
  unsigned short* wk_bf = (unsigned short*)(ws + off); off += WSZ;
  unsigned short* wv_bf = (unsigned short*)(ws + off); off += WSZ;
  unsigned short* wo_bf = (unsigned short*)(ws + off); off += WSZ;
  unsigned short* x_bf  = (unsigned short*)(ws + off); off += ACTSZ;
  unsigned short* Qp    = (unsigned short*)(ws + off); off += ACTSZ;
  unsigned short* KpT   = (unsigned short*)(ws + off); off += ACTSZ;
  unsigned short* VT    = (unsigned short*)(ws + off); off += ACTSZ;  // reused as attn
  float*          KV    = (float*)(ws + off);          off += (size_t)Bn * Hn * DH * DH * 4;
  unsigned short* KVT   = (unsigned short*)(ws + off); off += (size_t)Bn * Hn * DH * DH * 2;
  float*          Ksum  = (float*)(ws + off);          off += (size_t)Bn * Hn * DH * 4;
  unsigned short* attn  = VT;   // VT fully consumed before attn is produced

  // 1) convert weights + x to bf16 ; zero KV accumulator
  cvt_kernel<<<512,  256, 0, stream>>>(Wq, wq_bf);
  cvt_kernel<<<512,  256, 0, stream>>>(Wk, wk_bf);
  cvt_kernel<<<512,  256, 0, stream>>>(Wv, wv_bf);
  cvt_kernel<<<512,  256, 0, stream>>>(Wo, wo_bf);
  cvt_kernel<<<16384, 256, 0, stream>>>(x, x_bf);
  zero_kernel<<<1024, 256, 0, stream>>>(KV, Bn * Hn * DH * DH);

  // 2) fused projections + feature map (async double-buffered WMMA GEMMs)
  dim3 gproj(Mtot / 128, Dn / 128);
  gemm_kernel<MODE_Q><<<gproj, 256, 0, stream>>>(x_bf, wq_bf, bq, mask, Qp);
  gemm_kernel<MODE_K><<<gproj, 256, 0, stream>>>(x_bf, wk_bf, bk, mask, KpT);
  gemm_kernel<MODE_V><<<gproj, 256, 0, stream>>>(x_bf, wv_bf, bv, mask, VT);

  // 3) linear-attention state
  kv_kernel<<<dim3(Bn * Hn, 8), 256, 0, stream>>>(KpT, VT, KV);
  ksum_kernel<<<Bn * Hn, 256, 0, stream>>>(KpT, Ksum);
  kvt_kernel<<<(Bn * Hn * DH * DH) / 256, 256, 0, stream>>>(KV, KVT);

  // 4) numerator / denominator -> attn (bf16, (B,L,D))
  attn_kernel<<<dim3(Ln / 128, Bn * Hn), 256, 0, stream>>>(Qp, KVT, Ksum, attn);

  // 5) output projection -> d_out (fp32)
  gemm_kernel<MODE_O><<<gproj, 256, 0, stream>>>(attn, wo_bf, bo, nullptr, d_out);
}
